// SelfConv2d_21569325761153
// MI455X (gfx1250) — matrix operand order repaired and checked by reading, not compiled, not run on hardware
//
#include <hip/hip_runtime.h>
#include <hip/hip_bf16.h>

// ---------------------------------------------------------------------------
// SelfConv2d for MI455X (gfx1250, wave32, WMMA + async global->LDS).
// B=64 C=512 H=W=28 K=7 POOL=7 Cb=128. All GEMMs via v_wmma_f32_16x16x32_bf16.
// ---------------------------------------------------------------------------

typedef __attribute__((ext_vector_type(16))) __bf16 v16bf;
typedef __attribute__((ext_vector_type(8)))  float  v8f;
typedef __attribute__((ext_vector_type(4)))  int    v4i;

// Guarded CDNA5 async global->LDS path (ASYNCcnt, B128 granularity).
#if defined(__has_builtin)
#if __has_builtin(__builtin_amdgcn_global_load_async_to_lds_b128) && \
    __has_builtin(__builtin_amdgcn_s_wait_asynccnt)
#define HAVE_ASYNC_LDS 1
#endif
#endif

#if defined(HAVE_ASYNC_LDS)
// copies 16 bytes per active lane: LDS[dst] = GLOBAL[src]
// signature (from hipcc diagnostic): (v4i AS1* src, v4i AS3* dst, imm off, imm cpol)
#define ASYNC_CP16(dst_lds, src_glb)                                          \
    __builtin_amdgcn_global_load_async_to_lds_b128(                           \
        (__attribute__((address_space(1))) v4i*)(unsigned short*)(src_glb),   \
        (__attribute__((address_space(3))) v4i*)(dst_lds), 0, 0)
#define ASYNC_WAIT() __builtin_amdgcn_s_wait_asynccnt(0)
#endif

static __device__ __forceinline__ __bf16 us2bf(unsigned short u) {
    union { unsigned short u; __bf16 b; } x; x.u = u; return x.b;
}
static __device__ __forceinline__ unsigned short f2bfu(float f) {
    union { float f; unsigned u; } x; x.f = f;
    unsigned r = x.u + 0x7fffu + ((x.u >> 16) & 1u);   // round-to-nearest-even
    return (unsigned short)(r >> 16);
}
static __device__ __forceinline__ v8f v8zero() {
    v8f z; for (int i = 0; i < 8; ++i) z[i] = 0.f; return z;
}

// ---------------------------------------------------------------------------
// 0. zero fill for bf16 pad regions (Q/K p:49..63 pads must stay zero)
// ---------------------------------------------------------------------------
__global__ void fill_zero_u16(unsigned short* p, int n) {
    int i = blockIdx.x * 256 + threadIdx.x;
    if (i < n) p[i] = 0;
}

// fp32 -> bf16 copy (weights)
__global__ void cvt_bf16(const float* __restrict__ s, unsigned short* __restrict__ d, int n) {
    int i = blockIdx.x * 256 + threadIdx.x;
    if (i < n) d[i] = f2bfu(s[i]);
}

// Wkern 49x49 -> zero-padded 64x64 bf16
__global__ void cvt_wkern(const float* __restrict__ s, unsigned short* __restrict__ d) {
    int i = blockIdx.x * 256 + threadIdx.x;
    if (i < 64 * 64) {
        int r = i >> 6, c = i & 63;
        d[i] = (r < 49 && c < 49) ? f2bfu(s[r * 49 + c]) : (unsigned short)0;
    }
}

// ---------------------------------------------------------------------------
// 1. AdaptiveAvgPool(7,7) + BatchNorm(eval).  One block per (b,c).
//    Writes pf in two bf16 layouts:
//      PFb  [b*49+p][c]   (512 cols)  -> B operand of Q/K GEMM
//      PF2  [b*512+c][64] (p padded)  -> A operand of kw0 GEMM
// ---------------------------------------------------------------------------
__global__ void pool_bn(const float* __restrict__ x,
                        const float* __restrict__ gam, const float* __restrict__ bet,
                        const float* __restrict__ mu,  const float* __restrict__ var,
                        unsigned short* __restrict__ PFb,
                        unsigned short* __restrict__ PF2) {
    int bc = blockIdx.x;                 // 0..32767
    int b  = bc >> 9, c = bc & 511;
    int t  = threadIdx.x;                // 64 threads
    const float* xp = x + (size_t)bc * 784;
    if (t < 49) {
        int py = t / 7, px = t % 7;
        float s = 0.f;
        for (int dy = 0; dy < 4; ++dy)
            for (int dx = 0; dx < 4; ++dx)
                s += xp[(py * 4 + dy) * 28 + px * 4 + dx];
        float sc = gam[c] * rsqrtf(var[c] + 1e-5f);
        float v  = (s * (1.f / 16.f) - mu[c]) * sc + bet[c];
        unsigned short u = f2bfu(v);
        PFb[((size_t)(b * 49 + t) << 9) + c] = u;
        PF2[((size_t)bc << 6) + t]           = u;
    } else {
        PF2[((size_t)bc << 6) + t] = 0;      // pad p=49..63
    }
}

// ---------------------------------------------------------------------------
// 2. Generic bf16 GEMM-NT via WMMA:  C[m,n] = sum_k A[m,k] * B[n,k]
//    block = 128 threads (4 waves); each WG computes a 64x64 tile, each wave
//    16(M) x 64(N) via 4 f32 accumulators.  Kd multiple of 32.
//    Tiles staged global->LDS with GLOBAL_LOAD_ASYNC_TO_LDS_B128 when present.
//    mode 0: out[((n/49)*512+m)*64 + n%49] bf16   (Q/K, N=3136=49*64)
//    mode 1: out[m*64 + n] bf16                   (kw0 -> V)
// ---------------------------------------------------------------------------
__global__ void gemm_nt_bf16(const unsigned short* __restrict__ A,
                             const unsigned short* __restrict__ Bm,
                             unsigned short* __restrict__ out,
                             int Kd, int mode) {
    __shared__ __align__(16) unsigned short Al[64 * 40];   // pitch 40 us = 80 B
    __shared__ __align__(16) unsigned short Bl[64 * 40];
    const int tid  = threadIdx.x;
    const int wave = tid >> 5;
    const int lane = tid & 31;
    const int nn   = lane & 15;
    const int h    = lane >> 4;
    const int n0   = blockIdx.x * 64;
    const int m0   = blockIdx.y * 64;

    v8f acc[4];
    for (int g = 0; g < 4; ++g) acc[g] = v8zero();

    for (int k0 = 0; k0 < Kd; k0 += 32) {
        __syncthreads();
#if defined(HAVE_ASYNC_LDS)
        for (int idx = tid; idx < 256; idx += 128) {       // 64 rows x 4 b128
            int r = idx >> 2, c8 = (idx & 3) << 3;         // 8 ushorts / op
            ASYNC_CP16(&Al[r * 40 + c8], A  + (size_t)(m0 + r) * Kd + k0 + c8);
            ASYNC_CP16(&Bl[r * 40 + c8], Bm + (size_t)(n0 + r) * Kd + k0 + c8);
        }
        ASYNC_WAIT();
#else
        for (int idx = tid; idx < 64 * 32; idx += 128) {
            int r = idx >> 5, c = idx & 31;
            Al[r * 40 + c] = A[(size_t)(m0 + r) * Kd + k0 + c];
            Bl[r * 40 + c] = Bm[(size_t)(n0 + r) * Kd + k0 + c];
        }
#endif
        __syncthreads();
        // A fragment: 16x32 bf16 (ISA 7.12.2 layout)
        v16bf af;
        {
            int m = (wave << 4) + nn;
            for (int j = 0; j < 8; ++j) {
                int kb = ((j >> 2) << 4) + (h << 3) + ((j & 3) << 1);
                af[2 * j]     = us2bf(Al[m * 40 + kb]);
                af[2 * j + 1] = us2bf(Al[m * 40 + kb + 1]);
            }
        }
        for (int g = 0; g < 4; ++g) {
            v16bf bf;
            int n = (g << 4) + nn;
            for (int i = 0; i < 16; ++i)
                bf[i] = us2bf(Bl[n * 40 + ((i >> 3) << 4) + (h << 3) + (i & 7)]);
            acc[g] = __builtin_amdgcn_wmma_f32_16x16x32_bf16(
                false, af, false, bf, (short)0, acc[g], false, false);
        }
    }

    for (int g = 0; g < 4; ++g)
        for (int r = 0; r < 8; ++r) {
            int m = m0 + (wave << 4) + r + (h << 3);
            int n = n0 + (g << 4) + nn;
            float v = acc[g][r];
            if (mode == 0) {
                int b = n / 49, p = n % 49;
                out[((size_t)(b * 512 + m) << 6) + p] = f2bfu(v);
            } else {
                out[((size_t)m << 6) + n] = f2bfu(v);
            }
        }
}

// ---------------------------------------------------------------------------
// 3. Fused per-batch attention (flash style, online softmax):
//      S = Q[b] K[b]^T / 40 ; P = softmax_d(S) ; O = P V[b]
//    Q/K/V: bf16 [b][512][64] (p/kk padded with zeros).
//    Grid (4, 64): 128 Q-rows per WG, 8 waves x 16 rows; K/V streamed through
//    LDS in 128-row chunks (async-to-LDS B128 when available), d-tiles of 32;
//    P transposed through LDS to form the WMMA A operand.
// ---------------------------------------------------------------------------
__global__ void attn_fused(const unsigned short* __restrict__ Qg,
                           const unsigned short* __restrict__ Kg,
                           const unsigned short* __restrict__ Vg,
                           float* __restrict__ outKw) {
    __shared__ __align__(16) unsigned short Ql[128 * 72];  // pitch 72 us = 144 B
    __shared__ __align__(16) unsigned short Kl[128 * 72];
    __shared__ __align__(16) unsigned short Vl[128 * 72];
    __shared__ __align__(16) unsigned short Pl[8 * 16 * 34];  // per-wave 16x32 P

    const int b    = blockIdx.y;
    const int c0   = blockIdx.x * 128;
    const int tid  = threadIdx.x;                // 256
    const int wave = tid >> 5;
    const int lane = tid & 31;
    const int nn   = lane & 15;
    const int h    = lane >> 4;

    const unsigned short* Qb = Qg + ((size_t)b << 15);   // b*512*64
    const unsigned short* Kb = Kg + ((size_t)b << 15);
    const unsigned short* Vb = Vg + ((size_t)b << 15);

#if defined(HAVE_ASYNC_LDS)
    for (int idx = tid; idx < 128 * 8; idx += 256) {     // 128 rows x 8 b128
        int r = idx >> 3, c8 = (idx & 7) << 3;
        ASYNC_CP16(&Ql[r * 72 + c8], Qb + (((size_t)(c0 + r)) << 6) + c8);
    }
    ASYNC_WAIT();
#else
    for (int idx = tid; idx < 128 * 64; idx += 256) {
        int r = idx >> 6, c = idx & 63;
        Ql[r * 72 + c] = Qb[((size_t)(c0 + r) << 6) + c];
    }
#endif
    __syncthreads();

    // Q A-fragments for this wave's 16 rows (p 0..31 and 32..63)
    v16bf aq0, aq1;
    {
        int m = (wave << 4) + nn;
        for (int j = 0; j < 8; ++j) {
            int kb = ((j >> 2) << 4) + (h << 3) + ((j & 3) << 1);
            aq0[2 * j]     = us2bf(Ql[m * 72 + kb]);
            aq0[2 * j + 1] = us2bf(Ql[m * 72 + kb + 1]);
            aq1[2 * j]     = us2bf(Ql[m * 72 + 32 + kb]);
            aq1[2 * j + 1] = us2bf(Ql[m * 72 + 32 + kb + 1]);
        }
    }

    float mrow[8], lrow[8];
    v8f   O[4];
    for (int r = 0; r < 8; ++r) { mrow[r] = -1e30f; lrow[r] = 0.f; }
    for (int g = 0; g < 4; ++g) O[g] = v8zero();

    const float invT = 1.0f / 40.0f;

    for (int dc = 0; dc < 4; ++dc) {             // K/V chunks of 128 rows
        __syncthreads();
#if defined(HAVE_ASYNC_LDS)
        for (int idx = tid; idx < 128 * 8; idx += 256) {
            int r = idx >> 3, c8 = (idx & 7) << 3;
            ASYNC_CP16(&Kl[r * 72 + c8], Kb + (((size_t)(dc * 128 + r)) << 6) + c8);
            ASYNC_CP16(&Vl[r * 72 + c8], Vb + (((size_t)(dc * 128 + r)) << 6) + c8);
        }
        ASYNC_WAIT();
#else
        for (int idx = tid; idx < 128 * 64; idx += 256) {
            int r = idx >> 6, c = idx & 63;
            Kl[r * 72 + c] = Kb[((size_t)(dc * 128 + r) << 6) + c];
            Vl[r * 72 + c] = Vb[((size_t)(dc * 128 + r) << 6) + c];
        }
#endif
        __syncthreads();

        for (int dt = 0; dt < 4; ++dt) {         // d-tiles of 32 within chunk
            // ---- scores: S(16x32) = Q(16x64) . K_tile(32x64)^T / 40 ----
            v8f S0 = v8zero(), S1 = v8zero();
            {
                v16bf bk;
                int dl = (dt << 5) + nn;         // cols d 0..15 of tile
                for (int i = 0; i < 16; ++i) bk[i] = us2bf(Kl[dl * 72 + ((i >> 3) << 4) + (h << 3) + (i & 7)]);
                S0 = __builtin_amdgcn_wmma_f32_16x16x32_bf16(false, aq0, false, bk, (short)0, S0, false, false);
                for (int i = 0; i < 16; ++i) bk[i] = us2bf(Kl[dl * 72 + 32 + ((i >> 3) << 4) + (h << 3) + (i & 7)]);
                S0 = __builtin_amdgcn_wmma_f32_16x16x32_bf16(false, aq1, false, bk, (short)0, S0, false, false);
                int dl2 = (dt << 5) + 16 + nn;   // cols d 16..31 of tile
                for (int i = 0; i < 16; ++i) bk[i] = us2bf(Kl[dl2 * 72 + ((i >> 3) << 4) + (h << 3) + (i & 7)]);
                S1 = __builtin_amdgcn_wmma_f32_16x16x32_bf16(false, aq0, false, bk, (short)0, S1, false, false);
                for (int i = 0; i < 16; ++i) bk[i] = us2bf(Kl[dl2 * 72 + 32 + ((i >> 3) << 4) + (h << 3) + (i & 7)]);
                S1 = __builtin_amdgcn_wmma_f32_16x16x32_bf16(false, aq1, false, bk, (short)0, S1, false, false);
            }

            // ---- online softmax over the 32 columns of this tile ----
            float cf[8];
            for (int r = 0; r < 8; ++r) {
                float s0 = S0[r] * invT, s1 = S1[r] * invT;
                float rm = fmaxf(s0, s1);
                rm = fmaxf(rm, __shfl_xor(rm, 1, 16));
                rm = fmaxf(rm, __shfl_xor(rm, 2, 16));
                rm = fmaxf(rm, __shfl_xor(rm, 4, 16));
                rm = fmaxf(rm, __shfl_xor(rm, 8, 16));
                float mn = fmaxf(mrow[r], rm);
                float c_ = __expf(mrow[r] - mn);
                cf[r] = c_; mrow[r] = mn;
                float p0 = __expf(s0 - mn), p1 = __expf(s1 - mn);
                float rs = p0 + p1;
                rs += __shfl_xor(rs, 1, 16);
                rs += __shfl_xor(rs, 2, 16);
                rs += __shfl_xor(rs, 4, 16);
                rs += __shfl_xor(rs, 8, 16);
                lrow[r] = lrow[r] * c_ + rs;
                // stage P (bf16) for the lane->row transpose
                int pr = (r + (h << 3));
                Pl[wave * 544 + pr * 34 + nn]      = f2bfu(p0);
                Pl[wave * 544 + pr * 34 + 16 + nn] = f2bfu(p1);
            }
            for (int g = 0; g < 4; ++g)
                for (int r = 0; r < 8; ++r) O[g][r] *= cf[r];

            asm volatile("s_wait_dscnt 0" ::: "memory");  // in-wave LDS RAW

            // ---- P as A-fragment (16x32), accumulate O += P . V_tile ----
            v16bf ap;
            for (int j = 0; j < 8; ++j) {
                int kb = ((j >> 2) << 4) + (h << 3) + ((j & 3) << 1);
                ap[2 * j]     = us2bf(Pl[wave * 544 + nn * 34 + kb]);
                ap[2 * j + 1] = us2bf(Pl[wave * 544 + nn * 34 + kb + 1]);
            }
            for (int g = 0; g < 4; ++g) {
                v16bf bv;
                for (int i = 0; i < 16; ++i)
                    bv[i] = us2bf(Vl[((dt << 5) + ((i >> 3) << 4) + (h << 3) + (i & 7)) * 72 + (g << 4) + nn]);
                O[g] = __builtin_amdgcn_wmma_f32_16x16x32_bf16(
                    false, ap, false, bv, (short)0, O[g], false, false);
            }
        }
    }

    // ---- epilogue: O / l  ->  kw[b][c][49] fp32 ----
    for (int g = 0; g < 4; ++g)
        for (int r = 0; r < 8; ++r) {
            int kcol = (g << 4) + nn;
            if (kcol < 49) {
                int crow = c0 + (wave << 4) + r + (h << 3);
                outKw[((size_t)(b * 512 + crow)) * 49 + kcol] = O[g][r] / lrow[r];
            }
        }
}

// ---------------------------------------------------------------------------
// 4. fusion_conv0 (1x1, C->Cb, bias)
// ---------------------------------------------------------------------------
__global__ void fusion0(const float* __restrict__ Wf0, const float* __restrict__ bf0,
                        const float* __restrict__ kwA, float* __restrict__ F) {
    int idx = blockIdx.x * 256 + threadIdx.x;            // 64*128*49
    if (idx >= 64 * 128 * 49) return;
    int kk = idx % 49, o = (idx / 49) % 128, b = idx / (49 * 128);
    const float* wr = Wf0 + (size_t)o * 512;
    const float* kb = kwA + (size_t)b * 512 * 49 + kk;
    float acc = bf0[o];
    for (int c = 0; c < 512; ++c) {
        if ((c & 63) == 0) __builtin_prefetch(&kb[(size_t)(c + 64) * 49], 0, 0);
        acc += wr[c] * kb[(size_t)c * 49];
    }
    F[idx] = acc;
}

// ---------------------------------------------------------------------------
// 5. LayerNorm([Cb,K,K]) + tanh, per batch, in-place
// ---------------------------------------------------------------------------
__global__ void ln_tanh(float* __restrict__ F,
                        const float* __restrict__ lw, const float* __restrict__ lb) {
    __shared__ float r1[256], r2[256];
    __shared__ float s_mu, s_rs;
    int b = blockIdx.x, t = threadIdx.x;
    float* Fb = F + (size_t)b * 6272;
    float s = 0.f, s2 = 0.f;
    for (int i = t; i < 6272; i += 256) { float v = Fb[i]; s += v; s2 += v * v; }
    r1[t] = s; r2[t] = s2; __syncthreads();
    for (int w = 128; w > 0; w >>= 1) {
        if (t < w) { r1[t] += r1[t + w]; r2[t] += r2[t + w]; }
        __syncthreads();
    }
    if (t == 0) {
        float mu = r1[0] * (1.f / 6272.f);
        float vr = r2[0] * (1.f / 6272.f) - mu * mu;
        s_mu = mu; s_rs = rsqrtf(vr + 1e-5f);
    }
    __syncthreads();
    float mu = s_mu, rs = s_rs;
    for (int i = t; i < 6272; i += 256)
        Fb[i] = tanhf((Fb[i] - mu) * rs * lw[i] + lb[i]);
}

// ---------------------------------------------------------------------------
// 6. fusion_conv1 (1x1, Cb->C, bias)  ->  final dynamic kernels [b][c][49]
// ---------------------------------------------------------------------------
__global__ void fusion1(const float* __restrict__ Wf1, const float* __restrict__ bf1,
                        const float* __restrict__ F, float* __restrict__ KWf) {
    int idx = blockIdx.x * 256 + threadIdx.x;            // 64*512*49
    if (idx >= 64 * 512 * 49) return;
    int kk = idx % 49, o = (idx / 49) & 511, b = idx / (49 * 512);
    const float* wr = Wf1 + (size_t)o * 128;
    const float* fb = F + (size_t)b * 6272 + kk;
    float acc = bf1[o];
    __builtin_prefetch(&fb[64 * 49], 0, 0);
    for (int c = 0; c < 128; ++c) acc += wr[c] * fb[c * 49];
    KWf[idx] = acc;
}

// ---------------------------------------------------------------------------
// 7. Dynamic depthwise 7x7 conv, pad 3.  One block per (b,c); x plane tiled
//    into LDS with halo so each element is read once from HBM.
// ---------------------------------------------------------------------------
__global__ void dwconv(const float* __restrict__ x, const float* __restrict__ KWf,
                       float* __restrict__ out) {
    __shared__ float xs[34 * 34];
    __shared__ float ker[49];
    int bc = blockIdx.x, t = threadIdx.x;                // 256
    const float* xp = x + (size_t)bc * 784;
    for (int i = t; i < 34 * 34; i += 256) {
        int yy = i / 34 - 3, xx = i % 34 - 3;
        xs[i] = (yy >= 0 && yy < 28 && xx >= 0 && xx < 28) ? xp[yy * 28 + xx] : 0.f;
    }
    if (t < 49) ker[t] = KWf[(size_t)bc * 49 + t];
    __syncthreads();
    for (int p = t; p < 784; p += 256) {
        int oy = p / 28, ox = p % 28;
        float acc = 0.f;
        for (int dy = 0; dy < 7; ++dy)
            for (int dx = 0; dx < 7; ++dx)
                acc += xs[(oy + dy) * 34 + ox + dx] * ker[dy * 7 + dx];
        out[(size_t)bc * 784 + p] = acc;
    }
}

// ---------------------------------------------------------------------------
extern "C" void kernel_launch(void* const* d_in, const int* in_sizes, int n_in,
                              void* d_out, int out_size, void* d_ws, size_t ws_size,
                              hipStream_t stream) {
    const float* x    = (const float*)d_in[0];
    const float* gam  = (const float*)d_in[1];
    const float* bet  = (const float*)d_in[2];
    const float* bmu  = (const float*)d_in[3];
    const float* bvar = (const float*)d_in[4];
    const float* Wq   = (const float*)d_in[5];
    const float* Wk   = (const float*)d_in[6];
    const float* Wkr  = (const float*)d_in[7];
    const float* Wf0  = (const float*)d_in[8];
    const float* bf0  = (const float*)d_in[9];
    const float* lnw  = (const float*)d_in[10];
    const float* lnb  = (const float*)d_in[11];
    const float* Wf1  = (const float*)d_in[12];
    const float* bf1  = (const float*)d_in[13];

    char* ws = (char*)d_ws;
    size_t off = 0;
    auto take = [&](size_t bytes) { char* p = ws + off; off = (off + bytes + 255) & ~(size_t)255; return p; };

    unsigned short* PFb   = (unsigned short*)take((size_t)3136 * 512 * 2);   // [bp][c]
    unsigned short* PF2   = (unsigned short*)take((size_t)32768 * 64 * 2);   // [bc][64]
    unsigned short* WqB   = (unsigned short*)take((size_t)512 * 512 * 2);
    unsigned short* WkB   = (unsigned short*)take((size_t)512 * 512 * 2);
    unsigned short* WkrB  = (unsigned short*)take((size_t)64 * 64 * 2);
    unsigned short* Qb    = (unsigned short*)take((size_t)64 * 512 * 64 * 2);
    unsigned short* Kb    = (unsigned short*)take((size_t)64 * 512 * 64 * 2);
    unsigned short* Vb    = (unsigned short*)take((size_t)64 * 512 * 64 * 2);
    float*          kwA   = (float*)take((size_t)64 * 512 * 49 * 4);
    float*          Fbuf  = (float*)take((size_t)64 * 128 * 49 * 4);
    float*          KWfin = (float*)take((size_t)64 * 512 * 49 * 4);
    (void)ws_size; (void)in_sizes; (void)n_in; (void)out_size;

    // zero the Q/K bf16 buffers (p=49..63 pads must be zero for WMMA)
    {
        int n = 2 * 64 * 512 * 64;   // Qb+Kb contiguous region (ushorts)
        fill_zero_u16<<<(n + 255) / 256, 256, 0, stream>>>(Qb, n);
    }
    // weight conversions to bf16
    cvt_bf16<<<(512 * 512 + 255) / 256, 256, 0, stream>>>(Wq, WqB, 512 * 512);
    cvt_bf16<<<(512 * 512 + 255) / 256, 256, 0, stream>>>(Wk, WkB, 512 * 512);
    cvt_wkern<<<16, 256, 0, stream>>>(Wkr, WkrB);

    // pool + BN -> pf (both layouts, bf16)
    pool_bn<<<64 * 512, 64, 0, stream>>>(x, gam, bet, bmu, bvar, PFb, PF2);

    // Q = Wq @ PF, K = Wk @ PF   (M=512, N=3136, Kd=512)
    gemm_nt_bf16<<<dim3(49, 8), 128, 0, stream>>>(WqB, PFb, Qb, 512, 0);
    gemm_nt_bf16<<<dim3(49, 8), 128, 0, stream>>>(WkB, PFb, Kb, 512, 0);
    // kw0 = PF2 @ Wkern^T        (M=32768, N=64(pad), Kd=64(pad))
    gemm_nt_bf16<<<dim3(1, 512), 128, 0, stream>>>(PF2, WkrB, Vb, 64, 1);

    // fused attention: softmax(Q K^T / 40) @ kw0
    attn_fused<<<dim3(4, 64), 256, 0, stream>>>(Qb, Kb, Vb, kwA);

    // fusion convs + layernorm + tanh
    fusion0<<<(64 * 128 * 49 + 255) / 256, 256, 0, stream>>>(Wf0, bf0, kwA, Fbuf);
    ln_tanh<<<64, 256, 0, stream>>>(Fbuf, lnw, lnb);
    fusion1<<<(64 * 512 * 49 + 255) / 256, 256, 0, stream>>>(Wf1, bf1, Fbuf, KWfin);

    // dynamic depthwise conv
    dwconv<<<64 * 512, 256, 0, stream>>>(x, KWfin, (float*)d_out);
}
